// TransformerMultiViewFusion_14396730376656
// MI455X (gfx1250) — matrix operand order repaired and checked by reading, not compiled, not run on hardware
//
#include <hip/hip_runtime.h>

// ---------------------------------------------------------------------------
// Types for CDNA5 WMMA
// ---------------------------------------------------------------------------
typedef __attribute__((ext_vector_type(16))) __bf16 v16bf;
typedef __attribute__((ext_vector_type(8)))  float  v8f;
typedef __attribute__((ext_vector_type(4)))  unsigned int v4u;

union FragU { v4u u[2]; v16bf b; };

__device__ __forceinline__ unsigned short f2bf(float f) {
  unsigned u = __float_as_uint(f);
  u += 0x7fffu + ((u >> 16) & 1u);      // round-to-nearest-even
  return (unsigned short)(u >> 16);
}
__device__ __forceinline__ float bf2f(unsigned short h) {
  return __uint_as_float(((unsigned)h) << 16);
}

#define NTOK   73728      // B*H*W*NUM_CAMS = 2*96*96*4
#define NPIX   18432      // B*H*W
#define EDIM   256
#define FFDIM  1024
#define HWSZ   9216       // 96*96

// ---- LDS tiling for the GEMM (bytes) --------------------------------------
// A tile: 64 rows x 32 bf16 (64B) padded to 80B stride  -> 5120 B per stage
// B tile: 256 rows x 32 bf16 (64B) padded to 80B stride -> 20480 B per stage
#define ASTR   80
#define BSTR   80
#define A_ST0  0
#define A_ST1  5120
#define B_ST0  10240
#define B_ST1  30720
#define SMEM_BYTES 51200

// CDNA5 async global->LDS copy (16B per lane), tracked by ASYNCcnt
__device__ __forceinline__ void async_cp16(unsigned lds_byte_addr, const void* gp) {
  unsigned long long ga = (unsigned long long)(size_t)gp;
  asm volatile("global_load_async_to_lds_b128 %0, %1, off"
               :: "v"(lds_byte_addr), "v"(ga) : "memory");
}

// ---------------------------------------------------------------------------
// Weight f32 -> bf16 conversion
// ---------------------------------------------------------------------------
__global__ __launch_bounds__(256) void cvt_bf16_k(const float* __restrict__ src,
                                                  unsigned short* __restrict__ dst,
                                                  int n) {
  int i = blockIdx.x * 256 + threadIdx.x;
  if (i < n) dst[i] = f2bf(src[i]);
}

// ---------------------------------------------------------------------------
// Input projection: x[p*4+cam][e] = sum_c feat[cam][b][c][hw] * Wp[e][c] + bp[e]
// ---------------------------------------------------------------------------
__global__ __launch_bounds__(256) void proj_k(const float* __restrict__ feat,
                                              const float* __restrict__ Wp,
                                              const float* __restrict__ bp,
                                              float* __restrict__ x_f,
                                              unsigned short* __restrict__ x_b) {
  const int p  = blockIdx.x;
  const int e  = threadIdx.x;
  const int b  = p / HWSZ;
  const int hw = p - b * HWSZ;

  __shared__ float fsh[4][64];
  {
    int cam = threadIdx.x >> 6;
    int c   = threadIdx.x & 63;
    fsh[cam][c] = feat[(size_t)((cam * 2 + b) * 64 + c) * HWSZ + hw];
  }
  __syncthreads();

  float wrow[64];
#pragma unroll
  for (int c = 0; c < 64; ++c) wrow[c] = Wp[e * 64 + c];
  const float bias = bp[e];

#pragma unroll
  for (int cam = 0; cam < 4; ++cam) {
    float acc = bias;
#pragma unroll
    for (int c = 0; c < 64; ++c) acc += fsh[cam][c] * wrow[c];
    size_t idx = (size_t)(p * 4 + cam) * EDIM + e;
    x_f[idx] = acc;
    x_b[idx] = f2bf(acc);
  }
}

// ---------------------------------------------------------------------------
// bf16 WMMA GEMM:  C[M,N] = A[M,K] (row-major bf16) * W[N,K]^T + bias
// block = 8 waves; wave tile 32(M) x 64(N); block tile 64 x 256
// double-buffered LDS stages filled by global_load_async_to_lds_b128
// grid = (M/64, N/256)
// ---------------------------------------------------------------------------
template<int RELU, int OUTBF>
__global__ __launch_bounds__(256) void gemm_k(const unsigned short* __restrict__ A,
                                              const unsigned short* __restrict__ W,
                                              const float* __restrict__ bias,
                                              float* __restrict__ Cf,
                                              unsigned short* __restrict__ Cb,
                                              int K, int N) {
  __shared__ __align__(16) char smem[SMEM_BYTES];
  const int tid  = threadIdx.x;
  const int lane = tid & 31;
  const int wave = tid >> 5;
  const int wm   = wave >> 2;          // 0..1
  const int wn   = wave & 3;           // 0..3
  const int l15  = lane & 15;
  const int hi   = lane >> 4;
  const int mblk = blockIdx.x * 64;
  const int nblk = blockIdx.y * 256;
  const unsigned sbase = (unsigned)(size_t)(void*)smem;   // LDS byte offset

  const v8f vzero = {0.f, 0.f, 0.f, 0.f, 0.f, 0.f, 0.f, 0.f};
  v8f acc[2][4];
#pragma unroll
  for (int s = 0; s < 2; ++s)
#pragma unroll
    for (int j = 0; j < 4; ++j) acc[s][j] = vzero;

  // issue one stage of async copies: 1 A-chunk + 4 B-chunks per thread (16B each)
  auto issue = [&](int k0, int s) {
    const unsigned aB = sbase + (s ? A_ST1 : A_ST0);
    const unsigned bB = sbase + (s ? B_ST1 : B_ST0);
    {
      const int r = tid >> 2, c = tid & 3;   // 64 rows x 4 chunks
      async_cp16(aB + r * ASTR + c * 16, A + (size_t)(mblk + r) * K + k0 + c * 8);
    }
#pragma unroll
    for (int j = 0; j < 4; ++j) {            // 256 rows x 4 chunks
      const int g = j * 256 + tid;
      const int r = g >> 2, c = g & 3;
      async_cp16(bB + r * BSTR + c * 16, W + (size_t)(nblk + r) * K + k0 + c * 8);
    }
  };

  // load all 6 fragments from LDS up-front, then run 8 WMMAs (lets the
  // scheduler use partial s_wait_dscnt and hide LDS latency)
  auto compute = [&](int stg) {
    const unsigned aRel = stg ? A_ST1 : A_ST0;
    const unsigned bRel = stg ? B_ST1 : B_ST0;

    FragU afr[2];
#pragma unroll
    for (int s = 0; s < 2; ++s) {
      const unsigned off = aRel + (unsigned)(wm * 32 + s * 16 + l15) * ASTR + hi * 16;
      afr[s].u[0] = *(const v4u*)(smem + off);
      afr[s].u[1] = *(const v4u*)(smem + off + 32);
    }
    FragU bfr[4];
#pragma unroll
    for (int j = 0; j < 4; ++j) {
      const unsigned off = bRel + (unsigned)(wn * 64 + j * 16 + l15) * BSTR + hi * 16;
      bfr[j].u[0] = *(const v4u*)(smem + off);
      bfr[j].u[1] = *(const v4u*)(smem + off + 32);
    }
#pragma unroll
    for (int j = 0; j < 4; ++j) {
      acc[0][j] = __builtin_amdgcn_wmma_f32_16x16x32_bf16(false, afr[0].b, false, bfr[j].b,
                                                          (short)0, acc[0][j], false, false);
      acc[1][j] = __builtin_amdgcn_wmma_f32_16x16x32_bf16(false, afr[1].b, false, bfr[j].b,
                                                          (short)0, acc[1][j], false, false);
    }
  };

  // ---- software pipeline: steady-state (always prefetch next) + peeled tail
  issue(0, 0);
  int stage = 0;
  int k0 = 0;
  for (; k0 + 32 < K; k0 += 32) {
    issue(k0 + 32, stage ^ 1);
    asm volatile("s_wait_asynccnt 0x5" ::: "memory");   // oldest stage complete
    __syncthreads();
    compute(stage);
    __syncthreads();   // protect stage from being overwritten by next issue
    stage ^= 1;
  }
  asm volatile("s_wait_asynccnt 0x0" ::: "memory");
  __syncthreads();
  compute(stage);

  // C/D 16x16 f32 layout: lanes 0-15 -> col, rows +0..7 ; lanes 16-31 -> rows +8..15
  const int m0 = mblk + wm * 32;
  const int n0 = nblk + wn * 64;
#pragma unroll
  for (int s = 0; s < 2; ++s) {
    const int rbase = m0 + s * 16 + hi * 8;
#pragma unroll
    for (int j = 0; j < 4; ++j) {
      const int cn = n0 + j * 16 + l15;
      const float bia = bias[cn];
#pragma unroll
      for (int r = 0; r < 8; ++r) {
        float v = acc[s][j][r] + bia;
        if (RELU) v = fmaxf(v, 0.f);
        size_t idx = (size_t)(rbase + r) * N + cn;
        if (OUTBF) Cb[idx] = f2bf(v);
        else       Cf[idx] = v;
      }
    }
  }
}

// ---------------------------------------------------------------------------
// Attention over 4 cameras (seq-len 4), per (token, head) thread.
// qkv layout: [token][768] with q=0..255, k=256..511, v=512..767; head h -> h*32
// ---------------------------------------------------------------------------
__global__ __launch_bounds__(256) void attn_k(const unsigned short* __restrict__ qkv,
                                              unsigned short* __restrict__ sa) {
  const int tid   = blockIdx.x * 256 + threadIdx.x;  // 0 .. NTOK*8-1
  const int token = tid >> 3;
  const int head  = tid & 7;
  const int pixel = token >> 2;
  const float scale = 0.1767766952966369f;           // 1/sqrt(32)

  float q[32];
  const unsigned short* qp = qkv + (size_t)token * 768 + head * 32;
#pragma unroll
  for (int d = 0; d < 32; ++d) q[d] = bf2f(qp[d]);

  float s[4];
#pragma unroll
  for (int ck = 0; ck < 4; ++ck) {
    const unsigned short* kp = qkv + (size_t)(pixel * 4 + ck) * 768 + 256 + head * 32;
    float dot = 0.f;
#pragma unroll
    for (int d = 0; d < 32; ++d) dot += q[d] * bf2f(kp[d]);
    s[ck] = dot * scale;
  }
  float mx = fmaxf(fmaxf(s[0], s[1]), fmaxf(s[2], s[3]));
  float sum = 0.f;
#pragma unroll
  for (int ck = 0; ck < 4; ++ck) { s[ck] = __expf(s[ck] - mx); sum += s[ck]; }
  const float inv = 1.f / sum;

  float o[32];
#pragma unroll
  for (int d = 0; d < 32; ++d) o[d] = 0.f;
#pragma unroll
  for (int ck = 0; ck < 4; ++ck) {
    const float p = s[ck] * inv;
    const unsigned short* vp = qkv + (size_t)(pixel * 4 + ck) * 768 + 512 + head * 32;
#pragma unroll
    for (int d = 0; d < 32; ++d) o[d] += p * bf2f(vp[d]);
  }
  unsigned short* op = sa + (size_t)token * EDIM + head * 32;
#pragma unroll
  for (int d = 0; d < 32; ++d) op[d] = f2bf(o[d]);
}

// ---------------------------------------------------------------------------
// x = LayerNorm(x + y) * g + b ; writes f32 master and bf16 GEMM operand
// ---------------------------------------------------------------------------
__global__ __launch_bounds__(256) void add_ln_k(float* __restrict__ x,
                                                const float* __restrict__ y,
                                                const float* __restrict__ g,
                                                const float* __restrict__ b,
                                                unsigned short* __restrict__ xb) {
  const int t = blockIdx.x;
  const int e = threadIdx.x;
  const size_t idx = (size_t)t * EDIM + e;
  float v = x[idx] + y[idx];

  __shared__ float red[256];
  __shared__ float mu_s, rs_s;

  red[e] = v;
  __syncthreads();
#pragma unroll
  for (int s = 128; s > 0; s >>= 1) {
    if (e < s) red[e] += red[e + s];
    __syncthreads();
  }
  if (e == 0) mu_s = red[0] * (1.f / 256.f);
  __syncthreads();
  const float d = v - mu_s;
  red[e] = d * d;
  __syncthreads();
#pragma unroll
  for (int s = 128; s > 0; s >>= 1) {
    if (e < s) red[e] += red[e + s];
    __syncthreads();
  }
  if (e == 0) rs_s = rsqrtf(red[0] * (1.f / 256.f) + 1e-5f);
  __syncthreads();

  const float out = d * rs_s * g[e] + b[e];
  x[idx]  = out;
  xb[idx] = f2bf(out);
}

// ---------------------------------------------------------------------------
// camera-mean + 1x1 output conv
// ---------------------------------------------------------------------------
__global__ __launch_bounds__(256) void out_k(const float* __restrict__ x,
                                             const float* __restrict__ Wout,
                                             const float* __restrict__ bout,
                                             float* __restrict__ out) {
  const int p  = blockIdx.x;
  const int t  = threadIdx.x;
  const int b  = p / HWSZ;
  const int hw = p - b * HWSZ;

  __shared__ float fused[256];
  __shared__ float part[256];

  {
    const size_t base = (size_t)p * 4 * EDIM + t;
    fused[t] = 0.25f * (x[base] + x[base + EDIM] + x[base + 2 * EDIM] + x[base + 3 * EDIM]);
  }
  __syncthreads();

  const int co = t & 63;
  const int q  = t >> 6;
  float acc = 0.f;
#pragma unroll
  for (int j = 0; j < 64; ++j) {
    const int e = q * 64 + j;
    acc += fused[e] * Wout[co * EDIM + e];
  }
  part[t] = acc;
  __syncthreads();
  if (q == 0) {
    float r = bout[co] + part[co] + part[64 + co] + part[128 + co] + part[192 + co];
    out[(size_t)(b * 64 + co) * HWSZ + hw] = r;
  }
}

// ---------------------------------------------------------------------------
// Host launcher
// ---------------------------------------------------------------------------
extern "C" void kernel_launch(void* const* d_in, const int* in_sizes, int n_in,
                              void* d_out, int out_size, void* d_ws, size_t ws_size,
                              hipStream_t stream) {
  (void)in_sizes; (void)n_in; (void)out_size; (void)ws_size;

  const float* features = (const float*)d_in[0];
  const float* Wp    = (const float*)d_in[1];
  const float* bp    = (const float*)d_in[2];
  const float* Wqkv  = (const float*)d_in[3];
  const float* bqkv  = (const float*)d_in[4];
  const float* Wo    = (const float*)d_in[5];
  const float* bo    = (const float*)d_in[6];
  const float* W1    = (const float*)d_in[7];
  const float* b1    = (const float*)d_in[8];
  const float* W2    = (const float*)d_in[9];
  const float* b2    = (const float*)d_in[10];
  const float* g1    = (const float*)d_in[11];
  const float* be1   = (const float*)d_in[12];
  const float* g2    = (const float*)d_in[13];
  const float* be2   = (const float*)d_in[14];
  const float* Wout  = (const float*)d_in[15];
  const float* bout  = (const float*)d_in[16];
  float* out = (float*)d_out;

  // ---- workspace carve (256B aligned) ----
  char* base = (char*)d_ws;
  size_t off = 0;
  auto carve = [&](size_t bytes) -> void* {
    void* r = base + off;
    off = (off + bytes + 255) & ~(size_t)255;
    return r;
  };
  unsigned short* Wqkv_b = (unsigned short*)carve((size_t)2 * 768  * 256  * 2);
  unsigned short* Wo_b   = (unsigned short*)carve((size_t)2 * 256  * 256  * 2);
  unsigned short* W1_b   = (unsigned short*)carve((size_t)2 * 1024 * 256  * 2);
  unsigned short* W2_b   = (unsigned short*)carve((size_t)2 * 256  * 1024 * 2);
  float*          x_f    = (float*)         carve((size_t)NTOK * EDIM * 4);
  unsigned short* x_b    = (unsigned short*)carve((size_t)NTOK * EDIM * 2);
  unsigned short* h_b    = (unsigned short*)carve((size_t)NTOK * FFDIM * 2); // also qkv (768 cols)
  unsigned short* sa_b   = (unsigned short*)carve((size_t)NTOK * EDIM * 2);
  float*          tmp_f  = (float*)         carve((size_t)NTOK * EDIM * 4);

  // ---- weight conversion ----
  cvt_bf16_k<<<(2*768*256 + 255) / 256, 256, 0, stream>>>(Wqkv, Wqkv_b, 2*768*256);
  cvt_bf16_k<<<(2*256*256 + 255) / 256, 256, 0, stream>>>(Wo,   Wo_b,   2*256*256);
  cvt_bf16_k<<<(2*1024*256 + 255) / 256, 256, 0, stream>>>(W1,  W1_b,   2*1024*256);
  cvt_bf16_k<<<(2*256*1024 + 255) / 256, 256, 0, stream>>>(W2,  W2_b,   2*256*1024);

  // ---- input projection ----
  proj_k<<<NPIX, 256, 0, stream>>>(features, Wp, bp, x_f, x_b);

  const int MB = NTOK / 64;   // 1152 (block tile M=64)

  for (int i = 0; i < 2; ++i) {
    // QKV GEMM: [NTOK,256] x [768,256]^T -> bf16 qkv (stored in h_b, ld=768)
    gemm_k<0, 1><<<dim3(MB, 3), 256, 0, stream>>>(
        x_b, Wqkv_b + (size_t)i * 768 * 256, bqkv + i * 768, nullptr, h_b, 256, 768);

    // attention (4-camera softmax), -> sa_b
    attn_k<<<(NTOK * 8) / 256, 256, 0, stream>>>(h_b, sa_b);

    // Wo GEMM -> tmp_f (f32)
    gemm_k<0, 0><<<dim3(MB, 1), 256, 0, stream>>>(
        sa_b, Wo_b + (size_t)i * 256 * 256, bo + i * 256, tmp_f, nullptr, 256, 256);

    // x = LN(x + sa)
    add_ln_k<<<NTOK, 256, 0, stream>>>(x_f, tmp_f, g1 + i * 256, be1 + i * 256, x_b);

    // FFN up: relu(x @ W1^T + b1) -> h_b (bf16)
    gemm_k<1, 1><<<dim3(MB, 4), 256, 0, stream>>>(
        x_b, W1_b + (size_t)i * 1024 * 256, b1 + i * 1024, nullptr, h_b, 256, 1024);

    // FFN down: h @ W2^T + b2 -> tmp_f
    gemm_k<0, 0><<<dim3(MB, 1), 256, 0, stream>>>(
        h_b, W2_b + (size_t)i * 256 * 1024, b2 + i * 256, tmp_f, nullptr, 1024, 256);

    // x = LN(x + ff)
    add_ln_k<<<NTOK, 256, 0, stream>>>(x_f, tmp_f, g2 + i * 256, be2 + i * 256, x_b);
  }

  // camera mean + output 1x1 conv
  out_k<<<NPIX, 256, 0, stream>>>(x_f, Wout, bout, out);
}
